// DeepSeekMoE_12627203850990
// MI455X (gfx1250) — matrix operand order repaired and checked by reading, not compiled, not run on hardware
//
#include <hip/hip_runtime.h>
#include <hip/hip_bf16.h>
#include <math.h>

// Problem dims
constexpr int kH   = 2048;   // hidden
constexpr int kI   = 1408;   // routed expert intermediate
constexpr int kE   = 16;     // experts
constexpr int kT   = 1024;   // tokens (B*S)
constexpr int kTOPK = 4;
constexpr int kNG  = 4;      // groups
constexpr int kSI  = 2816;   // shared intermediate
constexpr float kSCALE = 2.5f;

constexpr int kLDSP = 40;    // padded LDS row stride (bf16 elems): 80B -> conflict-free b128 reads

typedef __attribute__((ext_vector_type(16))) __bf16 v16bf;
typedef __attribute__((ext_vector_type(8)))  __bf16 v8bf;
typedef __attribute__((ext_vector_type(8)))  float  v8f;

#define WMMA_BF16(a, b, c) \
  __builtin_amdgcn_wmma_f32_16x16x32_bf16(false, (a), false, (b), (short)0, (c), false, false)

// ---- B fragment: 32x16 bf16 from fp32 weight rows (B[k][n] = W[n][k]) ----
// lane L: n = L&15, hf = L>>4; VGPR j holds K = 8*hf + 2*j + (j>=4 ? 8 : 0) + {0,1}
__device__ __forceinline__ v16bf frag_b_f32(const float* row, int k0, int lane) {
  const int hf = (lane >> 4) & 1;
  v16bf f;
#pragma unroll
  for (int j = 0; j < 8; ++j) {
    const int K = (hf << 3) + (j << 1) + ((j >> 2) << 3);
    const float2 d = *(const float2*)(row + k0 + K);
    f[2 * j]     = (__bf16)d.x;
    f[2 * j + 1] = (__bf16)d.y;
  }
  return f;
}

// ---- A fragment: 16x32 bf16 from LDS-staged tile (row-major [32][kLDSP]) --
// lane L: m = mi*16 + (L&15), hf = L>>4; needs K = 8*hf+{0..7} and 16+8*hf+{0..7}
__device__ __forceinline__ v16bf frag_a_lds(const __bf16* tile, int mi, int lane) {
  const int hf = (lane >> 4) & 1;
  const __bf16* r = tile + (mi * 16 + (lane & 15)) * kLDSP;
  const v8bf lo = *(const v8bf*)(r + 8 * hf);        // ds_read_b128
  const v8bf hi = *(const v8bf*)(r + 16 + 8 * hf);   // ds_read_b128
  v16bf f;
#pragma unroll
  for (int i = 0; i < 8; ++i) { f[i] = lo[i]; f[8 + i] = hi[i]; }
  return f;
}

// ---- Gating: sigmoid scores, group-limited top-4, combine weights --------
__global__ void gate_topk_kernel(const float* __restrict__ x,
                                 const float* __restrict__ gw,
                                 const float* __restrict__ gb,
                                 float* __restrict__ combine,
                                 int* __restrict__ cnt,
                                 int* __restrict__ list) {
  const int t   = blockIdx.x;
  const int tid = threadIdx.x;          // 128 threads: 8 per expert
  const int e   = tid >> 3;
  const int sub = tid & 7;
  const float* xr = x + (size_t)t * kH;
  const float* wr = gw + (size_t)e * kH;
  float s = 0.f;
  for (int h = sub; h < kH; h += 8) s += xr[h] * wr[h];
#pragma unroll
  for (int o = 4; o; o >>= 1) s += __shfl_xor(s, o, 8);
  __shared__ float logit[kE];
  if (sub == 0) logit[e] = s;
  __syncthreads();
  if (tid == 0) {
    float sc[kE];
#pragma unroll
    for (int i = 0; i < kE; ++i) sc[i] = 1.f / (1.f + __expf(-logit[i])) + gb[i];
    float gs[kNG];
#pragma unroll
    for (int g = 0; g < kNG; ++g) {
      float m = sc[g * 4];
      for (int j = 1; j < 4; ++j) m = fmaxf(m, sc[g * 4 + j]);
      gs[g] = m;
    }
    int g0 = 0;
    for (int g = 1; g < kNG; ++g) if (gs[g] > gs[g0]) g0 = g;
    int g1 = -1;
    for (int g = 0; g < kNG; ++g) {
      if (g == g0) continue;
      if (g1 < 0 || gs[g] > gs[g1]) g1 = g;
    }
    float masked[kE];
#pragma unroll
    for (int i = 0; i < kE; ++i) {
      const int g = i >> 2;
      masked[i] = (g == g0 || g == g1) ? sc[i] : -1.f;   // real scores > 0
    }
    int isel[kTOPK]; float wsel[kTOPK]; float sum = 0.f;
    bool used[kE];
    for (int i = 0; i < kE; ++i) used[i] = false;
    for (int k = 0; k < kTOPK; ++k) {
      int best = -1;
      for (int i = 0; i < kE; ++i)
        if (!used[i] && (best < 0 || masked[i] > masked[best])) best = i;
      used[best] = true; isel[k] = best; wsel[k] = masked[best]; sum += masked[best];
    }
    const float inv = 1.f / (sum + 1e-6f);
    for (int i = 0; i < kE; ++i) combine[t * kE + i] = 0.f;
    for (int k = 0; k < kTOPK; ++k) {
      combine[t * kE + isel[k]] = wsel[k] * inv;
      const int pos = atomicAdd(&cnt[isel[k]], 1);
      list[isel[k] * kT + pos] = t;
    }
  }
}

// ---- Fused gate+up + SwiGLU. 4 waves/block share an LDS-staged A tile ----
// Block covers 32 tokens x 128 intermediate cols; wave w owns cols n0+w*32.
// list==nullptr -> identity token map (shared expert).
__global__ void __launch_bounds__(128)
up_kernel(const float* __restrict__ x,
          const float* __restrict__ wg,
          const float* __restrict__ wu,
          __bf16* __restrict__ hout,
          const int* __restrict__ list,
          const int* __restrict__ cnt,
          int Idim) {
  const int e  = blockIdx.z;
  const int nc = list ? cnt[e] : kT;
  const int m0 = blockIdx.y * 32;
  if (m0 >= nc) return;
  const int tid  = threadIdx.x;
  const int wave = tid >> 5;
  const int lane = tid & 31;
  const int n0   = blockIdx.x * 128 + wave * 32;

  __shared__ __align__(16) __bf16 atile[2][32 * kLDSP];

  // Staging role: thread copies row (tid>>2), K chunk (tid&3)*8 (fp32 -> bf16)
  const int srow = tid >> 2, skc = (tid & 3) * 8;
  int st;
  if (list) { const int* le = list + e * kT; st = le[min(m0 + srow, nc - 1)]; }
  else        st = m0 + srow;
  const float* sga = x + (size_t)st * kH + skc;
  __bf16* sds = &atile[0][0] + srow * kLDSP + skc;

  auto stage = [&](int buf, int k) {
    const float4 d0 = *(const float4*)(sga + k);
    const float4 d1 = *(const float4*)(sga + k + 4);
    v8bf v;
    v[0] = (__bf16)d0.x; v[1] = (__bf16)d0.y; v[2] = (__bf16)d0.z; v[3] = (__bf16)d0.w;
    v[4] = (__bf16)d1.x; v[5] = (__bf16)d1.y; v[6] = (__bf16)d1.z; v[7] = (__bf16)d1.w;
    *(v8bf*)(sds + buf * (32 * kLDSP)) = v;   // ds_store_b128
  };

  // Compute role: weight row pointers (fp32, cvt inline — the correct path,
  // TDM/async cannot convert so weights must pass through VGPRs anyway)
  const size_t wbase = (size_t)e * Idim * kH;
  const int nB = n0 + (lane & 15);
  const float* bg0 = wg + wbase + (size_t)nB * kH;
  const float* bg1 = bg0 + (size_t)16 * kH;
  const float* bu0 = wu + wbase + (size_t)nB * kH;
  const float* bu1 = bu0 + (size_t)16 * kH;

  v8f cg00 = {}, cg01 = {}, cg10 = {}, cg11 = {};
  v8f cu00 = {}, cu01 = {}, cu10 = {}, cu11 = {};

  stage(0, 0);
  __syncthreads();
  int p = 0;
  for (int k = 0; k < kH; k += 32) {
    const __bf16* tp = &atile[p][0];
    const v16bf A0  = frag_a_lds(tp, 0, lane);
    const v16bf A1  = frag_a_lds(tp, 1, lane);
    const v16bf Bg0 = frag_b_f32(bg0, k, lane);
    const v16bf Bg1 = frag_b_f32(bg1, k, lane);
    const v16bf Bu0 = frag_b_f32(bu0, k, lane);
    const v16bf Bu1 = frag_b_f32(bu1, k, lane);
    // run the weight stream ahead (global_prefetch_b8)
    __builtin_prefetch(bg0 + k + 256, 0, 1);
    __builtin_prefetch(bg1 + k + 256, 0, 1);
    __builtin_prefetch(bu0 + k + 256, 0, 1);
    __builtin_prefetch(bu1 + k + 256, 0, 1);
    if (k + 32 < kH) stage(p ^ 1, k + 32);
    cg00 = WMMA_BF16(A0, Bg0, cg00);
    cg01 = WMMA_BF16(A0, Bg1, cg01);
    cg10 = WMMA_BF16(A1, Bg0, cg10);
    cg11 = WMMA_BF16(A1, Bg1, cg11);
    cu00 = WMMA_BF16(A0, Bu0, cu00);
    cu01 = WMMA_BF16(A0, Bu1, cu01);
    cu10 = WMMA_BF16(A1, Bu0, cu10);
    cu11 = WMMA_BF16(A1, Bu1, cu11);
    __syncthreads();
    p ^= 1;
  }

  // h = silu(g) * u, store bf16.  C layout: n = lane&15, m = j + 8*(lane>>4).
  const int nl = lane & 15;
  const int hf = lane >> 4;
  const size_t hrowbase = list ? (size_t)e * kT : 0;
#pragma unroll
  for (int mi = 0; mi < 2; ++mi) {
#pragma unroll
    for (int j = 0; j < 8; ++j) {
      const int m = m0 + mi * 16 + hf * 8 + j;
      __bf16* hr = hout + (hrowbase + (size_t)m) * (size_t)Idim;
      float g = mi ? cg10[j] : cg00[j];
      float u = mi ? cu10[j] : cu00[j];
      float h = g / (1.f + __expf(-g)) * u;
      hr[n0 + nl] = (__bf16)h;
      g = mi ? cg11[j] : cg01[j];
      u = mi ? cu11[j] : cu01[j];
      h = g / (1.f + __expf(-g)) * u;
      hr[n0 + 16 + nl] = (__bf16)h;
    }
  }
}

// ---- Down projection. 4 waves/block, LDS-staged bf16 A tile --------------
// list==nullptr: shared expert -> plain store (initializes out).
// list!=nullptr: routed -> atomicAdd(out, combine*SCALE*val); slots >= cnt get 0.
__global__ void __launch_bounds__(128)
down_kernel(const __bf16* __restrict__ hbuf,
            const float* __restrict__ wd,
            const int* __restrict__ list,
            const int* __restrict__ cnt,
            const float* __restrict__ combine,
            float* __restrict__ out,
            int Kdim) {
  const int e  = blockIdx.z;
  const int nc = list ? cnt[e] : kT;
  const int m0 = blockIdx.y * 32;
  if (m0 >= nc) return;
  const int tid  = threadIdx.x;
  const int wave = tid >> 5;
  const int lane = tid & 31;
  const int n0   = blockIdx.x * 128 + wave * 32;   // H columns

  __shared__ __align__(16) __bf16 atile[2][32 * kLDSP];

  // Staging role: A is already bf16 -> straight 16B copies into LDS
  const int srow = tid >> 2, skc = (tid & 3) * 8;
  const size_t hbase = list ? (size_t)e * kT : 0;
  const __bf16* sga = hbuf + (hbase + (size_t)(m0 + srow)) * (size_t)Kdim + skc;
  __bf16* sds = &atile[0][0] + srow * kLDSP + skc;

  auto stage = [&](int buf, int k) {
    *(v8bf*)(sds + buf * (32 * kLDSP)) = *(const v8bf*)(sga + k);
  };

  const float* wde = wd + (size_t)e * kH * Kdim;
  const float* b0 = wde + (size_t)(n0 + (lane & 15)) * Kdim;
  const float* b1 = b0 + (size_t)16 * Kdim;

  v8f c00 = {}, c01 = {}, c10 = {}, c11 = {};

  stage(0, 0);
  __syncthreads();
  int p = 0;
  for (int k = 0; k < Kdim; k += 32) {
    const __bf16* tp = &atile[p][0];
    const v16bf A0 = frag_a_lds(tp, 0, lane);
    const v16bf A1 = frag_a_lds(tp, 1, lane);
    const v16bf B0 = frag_b_f32(b0, k, lane);
    const v16bf B1 = frag_b_f32(b1, k, lane);
    __builtin_prefetch(b0 + k + 256, 0, 1);
    __builtin_prefetch(b1 + k + 256, 0, 1);
    if (k + 32 < Kdim) stage(p ^ 1, k + 32);
    c00 = WMMA_BF16(A0, B0, c00);
    c01 = WMMA_BF16(A0, B1, c01);
    c10 = WMMA_BF16(A1, B0, c10);
    c11 = WMMA_BF16(A1, B1, c11);
    __syncthreads();
    p ^= 1;
  }

  const int nl = lane & 15;
  const int hf = lane >> 4;
  if (list) {
    const int* le = list + e * kT;
#pragma unroll
    for (int mi = 0; mi < 2; ++mi) {
#pragma unroll
      for (int j = 0; j < 8; ++j) {
        const int m = m0 + mi * 16 + hf * 8 + j;
        if (m < nc) {
          const int t = le[m];
          const float sc = combine[t * kE + e] * kSCALE;
          const float v0 = (mi ? c10[j] : c00[j]) * sc;
          const float v1 = (mi ? c11[j] : c01[j]) * sc;
          atomicAdd(&out[(size_t)t * kH + n0 + nl], v0);
          atomicAdd(&out[(size_t)t * kH + n0 + 16 + nl], v1);
        }
      }
    }
  } else {
#pragma unroll
    for (int mi = 0; mi < 2; ++mi) {
#pragma unroll
      for (int j = 0; j < 8; ++j) {
        const int m = m0 + mi * 16 + hf * 8 + j;
        out[(size_t)m * kH + n0 + nl]      = mi ? c10[j] : c00[j];
        out[(size_t)m * kH + n0 + 16 + nl] = mi ? c11[j] : c01[j];
      }
    }
  }
}

extern "C" void kernel_launch(void* const* d_in, const int* in_sizes, int n_in,
                              void* d_out, int out_size, void* d_ws, size_t ws_size,
                              hipStream_t stream) {
  (void)in_sizes; (void)n_in; (void)out_size; (void)ws_size;
  const float* x   = (const float*)d_in[0];  // [T,H]
  const float* gw  = (const float*)d_in[1];  // [E,H]
  const float* gb  = (const float*)d_in[2];  // [E]
  const float* wgt = (const float*)d_in[3];  // [E,I,H]
  const float* wup = (const float*)d_in[4];  // [E,I,H]
  const float* wdn = (const float*)d_in[5];  // [E,H,I]
  const float* swg = (const float*)d_in[6];  // [SI,H]
  const float* swu = (const float*)d_in[7];  // [SI,H]
  const float* swd = (const float*)d_in[8];  // [H,SI]
  float* out = (float*)d_out;                // [T,H]

  // Workspace carve-up (~52 MB total)
  char* ws = (char*)d_ws;
  size_t off = 0;
  auto carve = [&](size_t bytes) -> char* {
    char* p = ws + off;
    off = (off + bytes + 255) & ~(size_t)255;
    return p;
  };
  float*  combine = (float*)carve((size_t)kT * kE * sizeof(float));
  int*    cnt     = (int*)carve((size_t)kE * sizeof(int));
  int*    list    = (int*)carve((size_t)kE * kT * sizeof(int));
  __bf16* h_r     = (__bf16*)carve((size_t)kE * kT * kI * sizeof(__bf16));
  __bf16* h_s     = (__bf16*)carve((size_t)kT * kSI * sizeof(__bf16));

  hipMemsetAsync(cnt, 0, (size_t)kE * sizeof(int), stream);

  gate_topk_kernel<<<kT, 128, 0, stream>>>(x, gw, gb, combine, cnt, list);

  // Shared expert up (identity token map)
  up_kernel<<<dim3(kSI / 128, kT / 32, 1), 128, 0, stream>>>(
      x, swg, swu, h_s, nullptr, nullptr, kSI);
  // Routed experts up (sparse dispatch)
  up_kernel<<<dim3(kI / 128, kT / 32, kE), 128, 0, stream>>>(
      x, wgt, wup, h_r, list, cnt, kI);

  // Shared expert down: plain store, initializes all of out
  down_kernel<<<dim3(kH / 128, kT / 32, 1), 128, 0, stream>>>(
      h_s, swd, nullptr, nullptr, nullptr, out, kSI);
  // Routed experts down: scaled atomic accumulate
  down_kernel<<<dim3(kH / 128, kT / 32, kE), 128, 0, stream>>>(
      h_r, wdn, list, cnt, combine, out, kI);
}
